// ShoulderLSTM_29884382446303
// MI455X (gfx1250) — matrix operand order repaired and checked by reading, not compiled, not run on hardware
//
#include <hip/hip_runtime.h>
#include <hip/hip_fp16.h>

// CDNA5 wave32 WMMA types
typedef _Float16 v16h __attribute__((ext_vector_type(16)));
typedef float    v8f  __attribute__((ext_vector_type(8)));

#define TT   512      // sequence length
#define NB   512      // batch
#define HD   64       // hidden
#define RB   64       // batch rows per workgroup
#define HPAD 72       // padded f16 stride for 64-wide LDS tiles

__device__ __forceinline__ float sigf(float x) {
    return 1.0f / (1.0f + __expf(-x));
}
__device__ __forceinline__ float tanh_fast(float x) {
    x = fminf(fmaxf(x, -15.0f), 15.0f);
    float e = __expf(2.0f * x);
    return (e - 1.0f) / (e + 1.0f);
}

// A-fragment (16x32, f16) from row-major LDS tile, per ISA 7.12.2:
// lanes 0-15: M=lane, K in {0..7,16..23}; lanes 16-31: M=lane-16, K in {8..15,24..31}
// Two contiguous 8-halfword runs per lane -> compiler emits 2x ds_load_b128.
__device__ __forceinline__ v16h load_a(const _Float16* t, int stride, int mbase,
                                       int kbase, int lane) {
    int m  = mbase + (lane & 15);
    int kh = (lane >> 4) << 3;                 // 0 or 8
    const _Float16* row = t + m * stride + kbase + kh;
    v16h r;
#pragma unroll
    for (int j = 0; j < 8; ++j) {
        int k = ((j >> 2) << 4) + ((j & 3) << 1);   // {0,2,4,6,16,18,20,22}
        r[2 * j]     = row[k];
        r[2 * j + 1] = row[k + 1];
    }
    return r;
}

// B-fragment (32x16, f16) = W^T tile; W is row-major [256 x D] f32 in global.
// lanes 0-15: N=lane, K in {0..7,16..23}; lanes 16-31: N=lane-16, K in {8..15,24..31}.
__device__ __forceinline__ v16h load_b(const float* W, int D, int gbase,
                                       int kbase, int lane) {
    int n  = lane & 15;
    int kh = (lane >> 4) << 3;                 // 0 or 8
    const float* col = W + (size_t)(gbase + n) * D + kbase + kh;
    v16h r;
#pragma unroll
    for (int j = 0; j < 16; ++j) r[j] = (_Float16)col[j < 8 ? j : j + 8];
    return r;
}

__device__ __forceinline__ v8f wmma_f16(v16h a, v16h b, v8f c) {
    return __builtin_amdgcn_wmma_f32_16x16x32_f16(false, a, false, b,
                                                  (short)0, c, false, false);
}

// Persistent fused LSTM direction kernel, gates-in-registers variant.
// grid = (NB/RB, 2); blockIdx.y = direction (0=fwd, 1=bwd).
// 256 threads = 8 waves (2 per SIMD -> >=512 VGPRs/wave, no spills).
// wave w -> m-tile pair mh=(w>>2): m-tiles {2mh, 2mh+1}; gate block wq=(w&3).
// Wave owns n-tiles {wq, wq+4, wq+8, wq+12} == gate types i,f,g,o for hc in
// [16*wq, 16*wq+16): the LSTM cell is evaluated entirely from accumulators.
// DIN = 64 (layer0, fp32 input) or 128 (layer1, f16 input from workspace).
template <int DIN, bool WRITE_SEQ>
__global__ __launch_bounds__(256, 1) void lstm_dir_kernel(
    const float* __restrict__ xf32, const _Float16* __restrict__ xh16,
    const float* __restrict__ wih_f, const float* __restrict__ whh_f,
    const float* __restrict__ bih_f, const float* __restrict__ bhh_f,
    const float* __restrict__ wih_b, const float* __restrict__ whh_b,
    const float* __restrict__ bih_b, const float* __restrict__ bhh_b,
    _Float16* __restrict__ seq_out, float* __restrict__ last_out) {
    constexpr int XS   = DIN + 8;          // padded x-tile stride (f16)
    constexpr int NKX  = DIN / 32;         // K-chunks of input GEMM
    constexpr int XTSZ = RB * XS;          // one x-tile buffer (elements)

    const int tid  = threadIdx.x;
    const int lane = tid & 31;
    const int wave = tid >> 5;
    const int mh   = wave >> 2;            // m-tile pair 0..1
    const int wq   = wave & 3;             // hidden-column block 0..3
    const int dir  = blockIdx.y;
    const int rb0  = blockIdx.x * RB;

    const float* wih = dir ? wih_b : wih_f;
    const float* whh = dir ? whh_b : whh_f;
    const float* bih = dir ? bih_b : bih_f;
    const float* bhh = dir ? bhh_b : bhh_f;

    __shared__ _Float16 xt[2 * XTSZ];      // double-buffered staged x_t tile
    __shared__ _Float16 ht[RB * HPAD];     // recurrent h (f16)

    // ---- per-lane constants ----
    const int n  = lane & 15;
    const int mo = (lane >> 4) << 3;       // accumulator row block 0 or 8
    const int hc = wq * 16 + n;            // this lane's hidden column
    // per-gate bias, resident in registers for the whole sequence
    float bias_g[4];
#pragma unroll
    for (int q = 0; q < 4; ++q) bias_g[q] = bih[q * 64 + hc] + bhh[q * 64 + hc];

    // ---- zero h, stage x for t=0 ----
    for (int e = tid; e < RB * HPAD; e += 256) ht[e] = (_Float16)0.0f;
    {
        const int t0 = dir ? (TT - 1) : 0;
        if constexpr (DIN == 64) {
            for (int e = tid; e < RB * 64; e += 256) {
                int row = e >> 6, col = e & 63;
                xt[row * XS + col] =
                    (_Float16)xf32[((size_t)(rb0 + row) * TT + t0) * 64 + col];
            }
        } else {
            for (int e = tid; e < RB * 64; e += 256) {   // 64 uint pairs per row
                int row = e >> 6, col = (e & 63) * 2;
                *(unsigned int*)&xt[row * XS + col] =
                    *(const unsigned int*)&xh16[((size_t)(rb0 + row) * TT + t0) * 128 + col];
            }
        }
    }

    float creg[16], hreg[16];
#pragma unroll
    for (int r = 0; r < 16; ++r) { creg[r] = 0.0f; hreg[r] = 0.0f; }

    // ---- preload weight B-fragments (VGPR-resident across all steps) ----
    v16h bwih[4][NKX];
    v16h bwhh[4][2];
#pragma unroll
    for (int q = 0; q < 4; ++q) {
        const int gbase = (wq + 4 * q) * 16;
#pragma unroll
        for (int kc = 0; kc < NKX; ++kc)
            bwih[q][kc] = load_b(wih, DIN, gbase, kc * 32, lane);
#pragma unroll
        for (int kc = 0; kc < 2; ++kc)
            bwhh[q][kc] = load_b(whh, HD, gbase, kc * 32, lane);
    }

    __syncthreads();

    int buf = 0;
    for (int step = 0; step < TT; ++step) {
        const int t = dir ? (TT - 1 - step) : step;

        // ---- A-fragments: h_{t-1} and staged x_t, for both owned m-tiles ----
        const _Float16* xtile = xt + buf * XTSZ;
        v16h ax[2][NKX];
        v16h ah0[2], ah1[2];
#pragma unroll
        for (int im = 0; im < 2; ++im) {
            const int mt = 2 * mh + im;
#pragma unroll
            for (int kc = 0; kc < NKX; ++kc)
                ax[im][kc] = load_a(xtile, XS, mt * 16, kc * 32, lane);
            ah0[im] = load_a(ht, HPAD, mt * 16, 0, lane);
            ah1[im] = load_a(ht, HPAD, mt * 16, 32, lane);
        }

        // ---- issue global loads of x_{t+1} now; latency hides under WMMA ----
        float        xrf[16];
        unsigned int xrh[16];
        const bool more = (step + 1 < TT);
        const int  tn   = dir ? (TT - 2 - step) : (step + 1);
        if (more) {
            if constexpr (DIN == 64) {
#pragma unroll
                for (int e = 0; e < 16; ++e) {
                    int flat = e * 256 + tid;
                    int row = flat >> 6, col = flat & 63;
                    xrf[e] = xf32[((size_t)(rb0 + row) * TT + tn) * 64 + col];
                }
            } else {
#pragma unroll
                for (int e = 0; e < 16; ++e) {
                    int flat = e * 256 + tid;
                    int row = flat >> 6, col = (flat & 63) * 2;
                    xrh[e] = *(const unsigned int*)
                        &xh16[((size_t)(rb0 + row) * TT + tn) * 128 + col];
                }
            }
            // warm L2 two steps ahead (global_prefetch_b8)
            if (step + 2 < TT && tid < RB) {
                int t2 = dir ? (TT - 3 - step) : (step + 2);
                if constexpr (DIN == 64)
                    __builtin_prefetch(&xf32[((size_t)(rb0 + tid) * TT + t2) * 64], 0, 1);
                else
                    __builtin_prefetch(&xh16[((size_t)(rb0 + tid) * TT + t2) * 128], 0, 1);
            }
        }
        __syncthreads();   // A: everyone done reading ht / xt[buf]

        // ---- per m-tile: gate GEMM + LSTM cell entirely in registers ----
#pragma unroll
        for (int im = 0; im < 2; ++im) {
            const int mt = 2 * mh + im;
            v8f acc[4];
#pragma unroll
            for (int q = 0; q < 4; ++q) {
                v8f a = {0.f, 0.f, 0.f, 0.f, 0.f, 0.f, 0.f, 0.f};
#pragma unroll
                for (int kc = 0; kc < NKX; ++kc)
                    a = wmma_f16(ax[im][kc], bwih[q][kc], a);
                a = wmma_f16(ah0[im], bwhh[q][0], a);
                a = wmma_f16(ah1[im], bwhh[q][1], a);
                acc[q] = a;
            }
#pragma unroll
            for (int r = 0; r < 8; ++r) {
                int b = mt * 16 + mo + r;
                float gi = acc[0][r] + bias_g[0];
                float gf = acc[1][r] + bias_g[1];
                float gg = acc[2][r] + bias_g[2];
                float go = acc[3][r] + bias_g[3];
                float c  = sigf(gf) * creg[im * 8 + r] + sigf(gi) * tanh_fast(gg);
                float hn = sigf(go) * tanh_fast(c);
                creg[im * 8 + r] = c;
                hreg[im * 8 + r] = hn;
                ht[b * HPAD + hc] = (_Float16)hn;
                if constexpr (WRITE_SEQ)
                    seq_out[((size_t)(rb0 + b) * TT + t) * 128 + dir * 64 + hc] =
                        (_Float16)hn;
            }
        }

        // ---- commit staged x_{t+1} to the alternate LDS buffer ----
        if (more) {
            _Float16* xnext = xt + (buf ^ 1) * XTSZ;
            if constexpr (DIN == 64) {
#pragma unroll
                for (int e = 0; e < 16; ++e) {
                    int flat = e * 256 + tid;
                    int row = flat >> 6, col = flat & 63;
                    xnext[row * XS + col] = (_Float16)xrf[e];
                }
            } else {
#pragma unroll
                for (int e = 0; e < 16; ++e) {
                    int flat = e * 256 + tid;
                    int row = flat >> 6, col = (flat & 63) * 2;
                    *(unsigned int*)&xnext[row * XS + col] = xrh[e];
                }
            }
        }
        __syncthreads();   // B: h / x writes visible for next step
        buf ^= 1;
    }

    if constexpr (!WRITE_SEQ) {
#pragma unroll
        for (int im = 0; im < 2; ++im) {
#pragma unroll
            for (int r = 0; r < 8; ++r) {
                int b = (2 * mh + im) * 16 + mo + r;
                last_out[(size_t)(rb0 + b) * 128 + dir * 64 + hc] = hreg[im * 8 + r];
            }
        }
    }
}

// combined[B][128] -> relu(fc1) -> fc2 -> out[B]   (4 MFLOP, VALU is fine)
__global__ __launch_bounds__(256) void head_kernel(
    const float* __restrict__ comb, const float* __restrict__ fc1w,
    const float* __restrict__ fc1b, const float* __restrict__ fc2w,
    const float* __restrict__ fc2b, float* __restrict__ out) {
    int b = blockIdx.x * blockDim.x + threadIdx.x;
    if (b >= NB) return;
    const float* v = comb + (size_t)b * 128;
    float acc = fc2b[0];
    for (int j = 0; j < 32; ++j) {
        float s = fc1b[j];
        const float* w = fc1w + j * 128;
#pragma unroll 8
        for (int k = 0; k < 128; ++k) s += w[k] * v[k];
        acc += fc2w[j] * fmaxf(s, 0.0f);
    }
    out[b] = acc;
}

extern "C" void kernel_launch(void* const* d_in, const int* in_sizes, int n_in,
                              void* d_out, int out_size, void* d_ws, size_t ws_size,
                              hipStream_t stream) {
    (void)in_sizes; (void)n_in; (void)out_size; (void)ws_size;
    const float* x       = (const float*)d_in[0];
    const float* wih_l0f = (const float*)d_in[1];
    const float* whh_l0f = (const float*)d_in[2];
    const float* bih_l0f = (const float*)d_in[3];
    const float* bhh_l0f = (const float*)d_in[4];
    const float* wih_l0b = (const float*)d_in[5];
    const float* whh_l0b = (const float*)d_in[6];
    const float* bih_l0b = (const float*)d_in[7];
    const float* bhh_l0b = (const float*)d_in[8];
    const float* wih_l1f = (const float*)d_in[9];
    const float* whh_l1f = (const float*)d_in[10];
    const float* bih_l1f = (const float*)d_in[11];
    const float* bhh_l1f = (const float*)d_in[12];
    const float* wih_l1b = (const float*)d_in[13];
    const float* whh_l1b = (const float*)d_in[14];
    const float* bih_l1b = (const float*)d_in[15];
    const float* bhh_l1b = (const float*)d_in[16];
    const float* fc1w    = (const float*)d_in[17];
    const float* fc1b    = (const float*)d_in[18];
    const float* fc2w    = (const float*)d_in[19];
    const float* fc2b    = (const float*)d_in[20];

    // workspace: x1 [B][T][128] f16 (64 MB), then combined [B][128] f32 (256 KB)
    _Float16* x1  = (_Float16*)d_ws;
    float*   comb = (float*)((char*)d_ws + (size_t)NB * TT * 128 * sizeof(_Float16));

    dim3 grid(NB / RB, 2);  // 8 batch chunks x 2 directions

    lstm_dir_kernel<64, true><<<grid, 256, 0, stream>>>(
        x, nullptr,
        wih_l0f, whh_l0f, bih_l0f, bhh_l0f,
        wih_l0b, whh_l0b, bih_l0b, bhh_l0b,
        x1, nullptr);

    lstm_dir_kernel<128, false><<<grid, 256, 0, stream>>>(
        nullptr, x1,
        wih_l1f, whh_l1f, bih_l1f, bhh_l1f,
        wih_l1b, whh_l1b, bih_l1b, bhh_l1b,
        nullptr, comb);

    head_kernel<<<2, 256, 0, stream>>>(comb, fc1w, fc1b, fc2w, fc2b, (float*)d_out);
}
